// PointNetAE_10917806866867
// MI455X (gfx1250) — matrix operand order repaired and checked by reading, not compiled, not run on hardware
//
#include <hip/hip_runtime.h>
#include <hip/hip_bf16.h>

typedef __attribute__((ext_vector_type(16))) _Float16 v16h;
typedef __attribute__((ext_vector_type(8)))  float    v8f;

#define BATCH 8
#define KNBR 64

// ---- CDNA5 async global->LDS staging helpers (ASYNCcnt-tracked) ----------
__device__ __forceinline__ void async_ld_b32(void* lds, const void* g) {
  unsigned off = (unsigned)(unsigned long long)(uintptr_t)lds;  // LDS offset = low 32 bits
  asm volatile("global_load_async_to_lds_b32 %0, %1, off"
               :: "v"(off), "v"((unsigned long long)(uintptr_t)g) : "memory");
}
__device__ __forceinline__ void async_ld_b128(void* lds, const void* g) {
  unsigned off = (unsigned)(unsigned long long)(uintptr_t)lds;
  asm volatile("global_load_async_to_lds_b128 %0, %1, off"
               :: "v"(off), "v"((unsigned long long)(uintptr_t)g) : "memory");
}
__device__ __forceinline__ void wait_async0() {
  asm volatile("s_wait_asynccnt 0x0" ::: "memory");
}

// ---------------------------------------------------------------------------
// GEMM: OUT[R x N] = act(X[R x K] @ W[K x N] + bias[N])
// Block = 128 threads (4 waves), block tile 32 rows x 64 cols.
// Wave tile = 32 rows x 16 cols: 2 accumulators, 2x v_wmma_f32_16x16x32_f16
// per K-step sharing one B fragment. X/W tiles staged via async-to-LDS on the
// fast path; A fragments fed by ds b128 loads (row stride 36 floats keeps the
// 16-lane column access bank-conflict-free and rows 16B-aligned).
// ---------------------------------------------------------------------------
__global__ void gemm_bias_act(const float* __restrict__ X, const float* __restrict__ W,
                              const float* __restrict__ bias, float* __restrict__ OUT,
                              int R, int K, int N, int relu) {
  __shared__ __align__(16) float sX[32][36];   // 32 rows x 32 k (padded to 16B multiple)
  __shared__ __align__(16) float sW[32][64];   // 32 k x 64 cols (16B-aligned rows for b128)
  const int tid  = threadIdx.x;
  const int wave = tid >> 5;
  const int lane = tid & 31;
  const int lr   = lane & 15;
  const int hi   = lane >> 4;
  const int r0   = blockIdx.y * 32;
  const int c0   = blockIdx.x * 64;
  const int wc   = wave * 16;                  // wave's column offset in block tile
  const bool active = (c0 + wc) < N;

  v8f acc0 = {0.f,0.f,0.f,0.f,0.f,0.f,0.f,0.f};
  v8f acc1 = {0.f,0.f,0.f,0.f,0.f,0.f,0.f,0.f};

  for (int k0 = 0; k0 < K; k0 += 32) {
    const bool fullK = (k0 + 32) <= K;
    const bool asyncX = fullK && ((r0 + 32) <= R);
    const bool asyncW = fullK && ((c0 + 64) <= N) && ((N & 3) == 0);

    if (k0 + 32 < K) __builtin_prefetch(X + (size_t)r0 * K + (k0 + 32), 0, 1);

    // ---- stage X tile (32x32 f32) ----
    if (asyncX) {
      #pragma unroll
      for (int t = 0; t < 8; t++) {                  // 1024 elems / 128 thr
        int i = tid + t * 128;
        int rr = i >> 5, cc = i & 31;
        async_ld_b32(&sX[rr][cc], X + (size_t)(r0 + rr) * K + (k0 + cc));
      }
    } else {
      for (int i = tid; i < 32 * 32; i += 128) {
        int rr = i >> 5, cc = i & 31;
        int gr = r0 + rr, gc = k0 + cc;
        sX[rr][cc] = (gr < R && gc < K) ? X[(size_t)gr * K + gc] : 0.0f;
      }
    }
    // ---- stage W tile (32k x 64 cols f32) ----
    if (asyncW) {
      #pragma unroll
      for (int t = 0; t < 4; t++) {                  // 512 x b128 / 128 thr
        int i = tid + t * 128;
        int kk = i >> 4, cj = (i & 15) * 4;
        async_ld_b128(&sW[kk][cj], W + (size_t)(k0 + kk) * N + (c0 + cj));
      }
    } else {
      for (int i = tid; i < 32 * 64; i += 128) {
        int kk = i >> 6, cc = i & 63;
        int gk = k0 + kk, gc = c0 + cc;
        sW[kk][cc] = (gk < K && gc < N) ? W[(size_t)gk * N + gc] : 0.0f;
      }
    }
    if (asyncX || asyncW) wait_async0();
    __syncthreads();

    if (active) {
      v16h a0, a1, bfr;
      // A 16x32 layout per lane: two contiguous 8-float runs -> 4x b128 LDS loads
      {
        const float4* rp0 = (const float4*)(&sX[lr][hi * 8]);
        const float4* rp1 = (const float4*)(&sX[lr + 16][hi * 8]);
        float4 q0 = rp0[0], q1 = rp0[1], q2 = rp0[4], q3 = rp0[5];
        float4 u0 = rp1[0], u1 = rp1[1], u2 = rp1[4], u3 = rp1[5];
        a0[0]=(_Float16)q0.x; a0[1]=(_Float16)q0.y; a0[2]=(_Float16)q0.z; a0[3]=(_Float16)q0.w;
        a0[4]=(_Float16)q1.x; a0[5]=(_Float16)q1.y; a0[6]=(_Float16)q1.z; a0[7]=(_Float16)q1.w;
        a0[8]=(_Float16)q2.x; a0[9]=(_Float16)q2.y; a0[10]=(_Float16)q2.z; a0[11]=(_Float16)q2.w;
        a0[12]=(_Float16)q3.x; a0[13]=(_Float16)q3.y; a0[14]=(_Float16)q3.z; a0[15]=(_Float16)q3.w;
        a1[0]=(_Float16)u0.x; a1[1]=(_Float16)u0.y; a1[2]=(_Float16)u0.z; a1[3]=(_Float16)u0.w;
        a1[4]=(_Float16)u1.x; a1[5]=(_Float16)u1.y; a1[6]=(_Float16)u1.z; a1[7]=(_Float16)u1.w;
        a1[8]=(_Float16)u2.x; a1[9]=(_Float16)u2.y; a1[10]=(_Float16)u2.z; a1[11]=(_Float16)u2.w;
        a1[12]=(_Float16)u3.x; a1[13]=(_Float16)u3.y; a1[14]=(_Float16)u3.z; a1[15]=(_Float16)u3.w;
      }
      // B 32x16 layout: col = lane&15; lanes0-15 K=0..7,16..23, lanes16-31 K=8..15,24..31
      #pragma unroll
      for (int j = 0; j < 16; j++) {
        int kk = hi * 8 + (j & 7) + ((j >> 3) << 4);
        bfr[j] = (_Float16)sW[kk][wc + lr];
      }
      acc0 = __builtin_amdgcn_wmma_f32_16x16x32_f16(false, a0, false, bfr,
                                                    (short)0, acc0, false, false);
      acc1 = __builtin_amdgcn_wmma_f32_16x16x32_f16(false, a1, false, bfr,
                                                    (short)0, acc1, false, false);
    }
    __syncthreads();
  }

  if (active) {
    int cc = c0 + wc + lr;
    if (cc < N) {
      float bv = bias ? bias[cc] : 0.0f;
      if (r0 + 32 <= R) {
        // full row tile: branchless strided stores
        float* p0 = OUT + (size_t)(r0 + hi * 8) * N + cc;
        #pragma unroll
        for (int v = 0; v < 8; v++) {
          float v0 = acc0[v] + bv;
          float v1 = acc1[v] + bv;
          if (relu) { v0 = fmaxf(v0, 0.0f); v1 = fmaxf(v1, 0.0f); }
          p0[(size_t)v * N]        = v0;
          p0[(size_t)(v + 16) * N] = v1;
        }
      } else {
        #pragma unroll
        for (int v = 0; v < 8; v++) {
          int rr = r0 + v + hi * 8;               // C/D layout: M = v + hi*8
          if (rr < R) {
            float val = acc0[v] + bv;
            if (relu) val = fmaxf(val, 0.0f);
            OUT[(size_t)rr * N + cc] = val;
          }
          int rr1 = rr + 16;
          if (rr1 < R) {
            float val = acc1[v] + bv;
            if (relu) val = fmaxf(val, 0.0f);
            OUT[(size_t)rr1 * N + cc] = val;
          }
        }
      }
    }
  }
}

// ---------------------------------------------------------------------------
// Farthest point sampling: one block per batch, min-distance array in LDS.
// ---------------------------------------------------------------------------
__global__ void fps_kernel(const float* __restrict__ pos, int N, int M, int* __restrict__ idx_out) {
  __shared__ float mind[4096];
  __shared__ float rv[256];
  __shared__ int   ri[256];
  __shared__ float cur[3];
  const int b = blockIdx.x;
  const int t = threadIdx.x;
  const float* P = pos + (size_t)b * N * 3;
  int* idx = idx_out + (size_t)b * M;

  if (t == 0) { idx[0] = 0; cur[0] = P[0]; cur[1] = P[1]; cur[2] = P[2]; }
  __syncthreads();
  for (int p = t; p < N; p += 256) {
    float dx = P[p * 3] - cur[0], dy = P[p * 3 + 1] - cur[1], dz = P[p * 3 + 2] - cur[2];
    mind[p] = dx * dx + dy * dy + dz * dz;
  }
  __syncthreads();
  for (int m = 1; m < M; m++) {
    float bv = -1.0f; int bi = 0;
    for (int p = t; p < N; p += 256) {
      float v = mind[p];
      if (v > bv) { bv = v; bi = p; }
    }
    rv[t] = bv; ri[t] = bi;
    __syncthreads();
    for (int s = 128; s > 0; s >>= 1) {
      if (t < s) {
        if (rv[t + s] > rv[t] || (rv[t + s] == rv[t] && ri[t + s] < ri[t])) {
          rv[t] = rv[t + s]; ri[t] = ri[t + s];
        }
      }
      __syncthreads();
    }
    int sel = ri[0];
    if (t == 0) {
      idx[m] = sel;
      cur[0] = P[sel * 3]; cur[1] = P[sel * 3 + 1]; cur[2] = P[sel * 3 + 2];
    }
    __syncthreads();
    for (int p = t; p < N; p += 256) {
      float dx = P[p * 3] - cur[0], dy = P[p * 3 + 1] - cur[1], dz = P[p * 3 + 2] - cur[2];
      float v = dx * dx + dy * dy + dz * dz;
      if (v < mind[p]) mind[p] = v;
    }
    __syncthreads();
  }
}

__global__ void gather_ctr(const float* __restrict__ src, const int* __restrict__ idx,
                           float* __restrict__ dst, int Nsrc, int M) {
  int i = blockIdx.x * 256 + threadIdx.x;
  if (i >= BATCH * M) return;
  int b = i / M;
  int j = idx[i];
  const float* s = src + ((size_t)b * Nsrc + j) * 3;
  float* d = dst + (size_t)i * 3;
  d[0] = s[0]; d[1] = s[1]; d[2] = s[2];
}

// ---------------------------------------------------------------------------
// 64 nearest neighbors per center (iterative argmin over LDS distance array).
// ---------------------------------------------------------------------------
__global__ void knn_kernel(const float* __restrict__ pos, const float* __restrict__ ctr,
                           int N, int M, int* __restrict__ nbr, float* __restrict__ nd2) {
  __shared__ float d2s[4096];
  __shared__ float rv[128];
  __shared__ int   ri[128];
  const int c = blockIdx.x;            // b*M + m
  const int b = c / M;
  const int t = threadIdx.x;
  const float* P = pos + (size_t)b * N * 3;
  const float cx = ctr[(size_t)c * 3], cy = ctr[(size_t)c * 3 + 1], cz = ctr[(size_t)c * 3 + 2];

  for (int p = t; p < N; p += 128) {
    float dx = P[p * 3] - cx, dy = P[p * 3 + 1] - cy, dz = P[p * 3 + 2] - cz;
    d2s[p] = dx * dx + dy * dy + dz * dz;
  }
  __syncthreads();
  for (int k = 0; k < KNBR; k++) {
    float bv = 3.4e38f; int bi = N;
    for (int p = t; p < N; p += 128) {
      float v = d2s[p];
      if (v < bv || (v == bv && p < bi)) { bv = v; bi = p; }
    }
    rv[t] = bv; ri[t] = bi;
    __syncthreads();
    for (int s = 64; s > 0; s >>= 1) {
      if (t < s) {
        if (rv[t + s] < rv[t] || (rv[t + s] == rv[t] && ri[t + s] < ri[t])) {
          rv[t] = rv[t + s]; ri[t] = ri[t + s];
        }
      }
      __syncthreads();
    }
    if (t == 0) {
      nbr[(size_t)c * KNBR + k] = ri[0];
      nd2[(size_t)c * KNBR + k] = rv[0];
      d2s[ri[0]] = 3.4e38f;
    }
    __syncthreads();
  }
}

// feat row = [x_j (Cx), pos_j - pos_i (3)]
__global__ void gather_feat(const float* __restrict__ P, const float* __restrict__ X,
                            const float* __restrict__ ctr, const int* __restrict__ nbr,
                            float* __restrict__ feat, int M, int Cx) {
  int row = blockIdx.x * 256 + threadIdx.x;
  if (row >= M * KNBR) return;
  int m = row >> 6;
  int j = nbr[row];
  const float* c = ctr + (size_t)m * 3;
  float* f = feat + (size_t)row * (Cx + 3);
  for (int i = 0; i < Cx; i++) f[i] = X[(size_t)j * Cx + i];
  f[Cx]     = P[j * 3]     - c[0];
  f[Cx + 1] = P[j * 3 + 1] - c[1];
  f[Cx + 2] = P[j * 3 + 2] - c[2];
}

// masked max over the 64 neighbors (invalid -> -inf, self always valid)
__global__ void maskmax(const float* __restrict__ msg, const float* __restrict__ nd2,
                        float r2, float* __restrict__ out, int M, int H) {
  int i = blockIdx.x * 256 + threadIdx.x;
  if (i >= M * H) return;
  int m = i / H, h = i % H;
  float mx = -3.4e38f;
  for (int k = 0; k < KNBR; k++) {
    if (nd2[(size_t)m * KNBR + k] <= r2) {
      float v = msg[((size_t)m * KNBR + k) * H + h];
      mx = v > mx ? v : mx;
    }
  }
  out[(size_t)m * H + h] = mx;
}

__global__ void gpool(const float* __restrict__ x, float* __restrict__ pooled) {
  int i = blockIdx.x * 256 + threadIdx.x;          // 8*256
  if (i >= BATCH * 256) return;
  int b = i / 256, c = i % 256;
  const float* xb = x + (size_t)b * 128 * 256 + c;
  float mx = -3.4e38f;
  for (int m = 0; m < 128; m++) { float v = xb[(size_t)m * 256]; mx = v > mx ? v : mx; }
  pooled[i] = mx;
}

__global__ void zero_accum(float* a) { if (threadIdx.x < 8) a[threadIdx.x] = 0.0f; }

// mu / logvar / reparam + KL accumulation
__global__ void mu_logvar_z(const float* __restrict__ pooled,
                            const float* __restrict__ Wm, const float* __restrict__ bm,
                            const float* __restrict__ Wl, const float* __restrict__ bl,
                            const float* __restrict__ eps, float* __restrict__ z,
                            float* __restrict__ accum) {
  int i = blockIdx.x * 256 + threadIdx.x;
  if (i >= BATCH * 64) return;
  int b = i / 64, j = i % 64;
  const float* p = pooled + (size_t)b * 256;
  float mu = bm[j], lv = bl[j];
  for (int k = 0; k < 256; k++) {
    float pv = p[k];
    mu += pv * Wm[(size_t)k * 64 + j];
    lv += pv * Wl[(size_t)k * 64 + j];
  }
  z[i] = mu + expf(0.5f * lv) * eps[i];
  float kt = 1.0f + lv - mu * mu - expf(lv);
  atomicAdd(&accum[0], kt);
}

__global__ void fold1_in(const float* __restrict__ z, float* __restrict__ out) {
  int row = blockIdx.x * 256 + threadIdx.x;
  if (row >= BATCH * 2500) return;
  int b = row / 2500, g = row % 2500;
  const float* zb = z + (size_t)b * 64;
  float* f = out + (size_t)row * 66;
  for (int i = 0; i < 64; i++) f[i] = zb[i];
  int gi = g / 50, gj = g % 50;
  f[64] = -0.3f + 0.6f * (float)gi / 49.0f;
  f[65] = -0.3f + 0.6f * (float)gj / 49.0f;
}

__global__ void fold2_in(const float* __restrict__ z, const float* __restrict__ r1,
                         float* __restrict__ out) {
  int row = blockIdx.x * 256 + threadIdx.x;
  if (row >= BATCH * 2500) return;
  int b = row / 2500;
  const float* zb = z + (size_t)b * 64;
  float* f = out + (size_t)row * 67;
  for (int i = 0; i < 64; i++) f[i] = zb[i];
  f[64] = r1[(size_t)row * 3];
  f[65] = r1[(size_t)row * 3 + 1];
  f[66] = r1[(size_t)row * 3 + 2];
}

// wave-per-point min distance to the opposing cloud
__global__ void chamfer_min(const float* __restrict__ A, int nA,
                            const float* __restrict__ Bp, int nB,
                            float* __restrict__ mins) {
  int wid  = (blockIdx.x * 256 + threadIdx.x) >> 5;
  int lane = threadIdx.x & 31;
  if (wid >= BATCH * nA) return;
  int b = wid / nA, n = wid % nA;
  const float* a = A + ((size_t)b * nA + n) * 3;
  float ax = a[0], ay = a[1], az = a[2];
  const float* Bb = Bp + (size_t)b * nB * 3;
  float mn = 3.4e38f;
  for (int m = lane; m < nB; m += 32) {
    float dx = ax - Bb[m * 3], dy = ay - Bb[m * 3 + 1], dz = az - Bb[m * 3 + 2];
    float d2 = dx * dx + dy * dy + dz * dz;
    mn = d2 < mn ? d2 : mn;
  }
  for (int off = 16; off > 0; off >>= 1) {
    float o = __shfl_xor(mn, off, 32);
    mn = o < mn ? o : mn;
  }
  if (lane == 0) mins[wid] = sqrtf(fmaxf(mn, 1e-12f));
}

__global__ void finalize(const float* __restrict__ minA, const float* __restrict__ minB,
                         const float* __restrict__ accum, float* __restrict__ out) {
  __shared__ float sA[256], sB[256];
  int t = threadIdx.x;
  float a = 0.0f, b = 0.0f;
  for (int i = t; i < BATCH * 2500; i += 256) a += minA[i];
  for (int i = t; i < BATCH * 4096; i += 256) b += minB[i];
  sA[t] = a; sB[t] = b;
  __syncthreads();
  for (int s = 128; s > 0; s >>= 1) {
    if (t < s) { sA[t] += sA[t + s]; sB[t] += sB[t + s]; }
    __syncthreads();
  }
  if (t == 0) {
    float chamfer = sA[0] / (float)(BATCH * 2500) + sB[0] / (float)(BATCH * 4096);
    float kl = -0.5f * (accum[0] / (float)BATCH);
    out[0] = chamfer + 1e-3f * kl;
  }
}

// ---------------------------------------------------------------------------
extern "C" void kernel_launch(void* const* d_in, const int* in_sizes, int n_in,
                              void* d_out, int out_size, void* d_ws, size_t ws_size,
                              hipStream_t stream) {
  (void)in_sizes; (void)n_in; (void)out_size; (void)ws_size;
  auto F = [&](int i) { return (const float*)d_in[i]; };
  const float* pos = F(0);
  const float* eps = F(1);
  // params: sa1(2..5) sa2(6..9) sa3(10..13) fc_mu(14,15) fc_logvar(16,17)
  //         fold1(18..23) fold2(24..29)

  char* wsp = (char*)d_ws;
  size_t off = 0;
  auto alloc = [&](size_t bytes) -> void* {
    void* p = (void*)(wsp + off);
    off += (bytes + 255) & ~(size_t)255;
    return p;
  };

  int*   idx1 = (int*)alloc(sizeof(int) * BATCH * 2048);
  int*   idx2 = (int*)alloc(sizeof(int) * BATCH * 512);
  int*   idx3 = (int*)alloc(sizeof(int) * BATCH * 128);
  float* pos1 = (float*)alloc(sizeof(float) * BATCH * 2048 * 3);
  float* pos2 = (float*)alloc(sizeof(float) * BATCH * 512 * 3);
  float* pos3 = (float*)alloc(sizeof(float) * BATCH * 128 * 3);
  int*   nbr1 = (int*)alloc(sizeof(int) * BATCH * 2048 * KNBR);
  int*   nbr2 = (int*)alloc(sizeof(int) * BATCH * 512 * KNBR);
  int*   nbr3 = (int*)alloc(sizeof(int) * BATCH * 128 * KNBR);
  float* d21  = (float*)alloc(sizeof(float) * BATCH * 2048 * KNBR);
  float* d22  = (float*)alloc(sizeof(float) * BATCH * 512 * KNBR);
  float* d23  = (float*)alloc(sizeof(float) * BATCH * 128 * KNBR);
  float* x1   = (float*)alloc(sizeof(float) * BATCH * 2048 * 64);
  float* x2   = (float*)alloc(sizeof(float) * BATCH * 512 * 128);
  float* x3   = (float*)alloc(sizeof(float) * BATCH * 128 * 256);
  float* pooled = (float*)alloc(sizeof(float) * BATCH * 256);
  float* zbuf   = (float*)alloc(sizeof(float) * BATCH * 64);
  float* accum  = (float*)alloc(sizeof(float) * 8);
  float* recon1 = (float*)alloc(sizeof(float) * BATCH * 2500 * 3);
  float* recon  = (float*)alloc(sizeof(float) * BATCH * 2500 * 3);
  float* minA   = (float*)alloc(sizeof(float) * BATCH * 2500);
  float* minB   = (float*)alloc(sizeof(float) * BATCH * 4096);
  float* bufA   = (float*)alloc(sizeof(float) * 2195456);   // max row-feature buffer
  float* bufB   = (float*)alloc(sizeof(float) * 10240000);  // max hidden buffer
  float* bufC   = (float*)alloc(sizeof(float) * 10240000);

  auto gemm = [&](const float* X, const float* W, const float* bias, float* OUT,
                  int R, int K, int N, int relu) {
    dim3 grid((N + 63) / 64, (R + 31) / 32);
    gemm_bias_act<<<grid, 128, 0, stream>>>(X, W, bias, OUT, R, K, N, relu);
  };

  // ---- Set Abstraction levels ----
  const int   Nsrc[3] = {4096, 2048, 512};
  const int   Mlvl[3] = {2048, 512, 128};
  const float r2l[3]  = {0.04f, 0.16f, 0.64f};
  const int   Cxl[3]  = {0, 64, 128};
  const int   Hl[3]   = {64, 128, 256};
  const float* srcPos[3] = {pos, pos1, pos2};
  const float* srcX[3]   = {nullptr, x1, x2};
  int*   idxA[3] = {idx1, idx2, idx3};
  float* ctrA[3] = {pos1, pos2, pos3};
  int*   nbrA[3] = {nbr1, nbr2, nbr3};
  float* d2A[3]  = {d21, d22, d23};
  float* xA[3]   = {x1, x2, x3};
  const float* W0a[3] = {F(2), F(6), F(10)};
  const float* b0a[3] = {F(3), F(7), F(11)};
  const float* W1a[3] = {F(4), F(8), F(12)};
  const float* b1a[3] = {F(5), F(9), F(13)};

  for (int l = 0; l < 3; l++) {
    int N = Nsrc[l], M = Mlvl[l], cx = Cxl[l], h = Hl[l];
    fps_kernel<<<BATCH, 256, 0, stream>>>(srcPos[l], N, M, idxA[l]);
    gather_ctr<<<(BATCH * M + 255) / 256, 256, 0, stream>>>(srcPos[l], idxA[l], ctrA[l], N, M);
    knn_kernel<<<BATCH * M, 128, 0, stream>>>(srcPos[l], ctrA[l], N, M, nbrA[l], d2A[l]);
    int rows = M * KNBR;
    for (int b = 0; b < BATCH; b++) {
      gather_feat<<<(rows + 255) / 256, 256, 0, stream>>>(
          srcPos[l] + (size_t)b * N * 3,
          srcX[l] ? srcX[l] + (size_t)b * N * cx : nullptr,
          ctrA[l] + (size_t)b * M * 3,
          nbrA[l] + (size_t)b * M * KNBR, bufA, M, cx);
      gemm(bufA, W0a[l], b0a[l], bufB, rows, cx + 3, h, 1);
      gemm(bufB, W1a[l], b1a[l], bufC, rows, h, h, 0);
      maskmax<<<(M * h + 255) / 256, 256, 0, stream>>>(
          bufC, d2A[l] + (size_t)b * M * KNBR, r2l[l],
          xA[l] + (size_t)b * M * h, M, h);
    }
  }

  // ---- encoder head ----
  gpool<<<(BATCH * 256 + 255) / 256, 256, 0, stream>>>(x3, pooled);
  zero_accum<<<1, 32, 0, stream>>>(accum);
  mu_logvar_z<<<(BATCH * 64 + 255) / 256, 256, 0, stream>>>(
      pooled, F(14), F(15), F(16), F(17), eps, zbuf, accum);

  // ---- folding decoder ----
  const int G = BATCH * 2500;
  fold1_in<<<(G + 255) / 256, 256, 0, stream>>>(zbuf, bufA);
  gemm(bufA, F(18), F(19), bufB, G, 66, 512, 1);
  gemm(bufB, F(20), F(21), bufC, G, 512, 512, 1);
  gemm(bufC, F(22), F(23), recon1, G, 512, 3, 0);
  fold2_in<<<(G + 255) / 256, 256, 0, stream>>>(zbuf, recon1, bufA);
  gemm(bufA, F(24), F(25), bufB, G, 67, 512, 1);
  gemm(bufB, F(26), F(27), bufC, G, 512, 512, 1);
  gemm(bufC, F(28), F(29), recon, G, 512, 3, 0);

  // ---- chamfer + loss ----
  chamfer_min<<<(BATCH * 2500 * 32 + 255) / 256, 256, 0, stream>>>(recon, 2500, pos, 4096, minA);
  chamfer_min<<<(BATCH * 4096 * 32 + 255) / 256, 256, 0, stream>>>(pos, 4096, recon, 2500, minB);
  finalize<<<1, 256, 0, stream>>>(minA, minB, accum, (float*)d_out);
}
